// SpatioTemporalModel_15109694947637
// MI455X (gfx1250) — matrix operand order repaired and checked by reading, not compiled, not run on hardware
//
#include <hip/hip_runtime.h>

#define NNODES 200
#define SEQ_S  256   // attention sequence length (= BATCH)
#define DMODEL 128
#define HEADS_N 4
#define DHEAD  32
#define NLAYERS 2
#define FOUT 64
#define GOUT 64
#define NEDGE 1600
#define ETOT (NEDGE + NNODES)      // 1800 (edges + self loops)
#define NROWS (SEQ_S * NNODES)     // 51200 flat (node,batch) rows

typedef __attribute__((ext_vector_type(16))) _Float16 v16h;
typedef __attribute__((ext_vector_type(8)))  _Float16 v8h;
typedef __attribute__((ext_vector_type(8)))  float    v8f;

// ---------------------------------------------------------------------------
// WMMA fragment helpers (wave32; A/B use this code's contiguous k order, C/D per CDNA5 ISA §7.12.2)
// A 16x32 f16: lane m (0-15) holds row m, K = 0..15 contiguous; lane m+16 holds
//              row m, K = 16..31           -> two contiguous 16B loads.
// B 32x16 f16: lane n (0-15) holds col n, K = 0..15 contiguous; lane n+16
//              holds col n, K = 16..31     -> one contiguous 32B load.
//              Column n of B == row n of the [out,in] weight matrix.
// C/D 16x16 f32: lane L: col = L&15, rows = 8*(L>>4) + 0..7 over 8 VGPRs.
// ---------------------------------------------------------------------------
__device__ __forceinline__ v16h load_a_frag(const _Float16* __restrict__ A, int lda,
                                            int m0, int k0, int lane, int M) {
  int m = m0 + (lane & 15);
  if (m >= M) m = M - 1;                 // clamp (stores are guarded)
  int koff = (lane >> 4) << 4;           // 0 or 16
  const _Float16* p = A + (size_t)m * lda + k0 + koff;
  v8h lo = *(const v8h*)(p);
  v8h hi = *(const v8h*)(p + 8);
  v16h f;
#pragma unroll
  for (int i = 0; i < 8; ++i) { f[i] = lo[i]; f[i + 8] = hi[i]; }
  return f;
}

__device__ __forceinline__ v16h load_b_frag(const _Float16* __restrict__ Wb, int ldw,
                                            int k0, int lane, int nRem) {
  int n = lane & 15;
  if (n >= nRem) n = nRem - 1;
  int koff = (lane >> 4) << 4;           // 0 or 16
  return *(const v16h*)(Wb + (size_t)n * ldw + k0 + koff);
}

// ---------------------------------------------------------------------------
// Generic batched GEMM:  Out[node] = act( A[node] (MxK,f16) @ W[node]^T + b )
// W stored [Nout x K] row-major (torch Linear layout). Block = 8 waves laid
// out 2(M) x 4(N); each wave owns one 16x16 tile; K-loop of 16x16x32 WMMAs.
// ---------------------------------------------------------------------------
template <bool RELU, bool OUT_HALF, bool HAS_BIAS>
__global__ __launch_bounds__(256) void gemm_node(
    const _Float16* __restrict__ A, long aStride, int lda,
    const _Float16* __restrict__ W, long wStride, int ldw,
    const float* __restrict__ bias, long bStride,
    void* __restrict__ Out, long oStride, int ldo,
    int M, int Nout, int K) {
  int node = blockIdx.x;
  int wave = threadIdx.x >> 5, lane = threadIdx.x & 31;
  int mt = blockIdx.y * 2 + (wave >> 2);
  int nt = blockIdx.z * 4 + (wave & 3);
  int m0 = mt * 16, n0 = nt * 16;
  if (n0 >= Nout || m0 >= M + 15) return;   // wave-uniform, no barriers used
  const _Float16* a_base = A + (size_t)node * aStride;
  const _Float16* w_base = W + (size_t)node * wStride + (size_t)n0 * ldw;
  v8f acc = {};
  for (int k0 = 0; k0 < K; k0 += 32) {
    v16h af = load_a_frag(a_base, lda, m0, k0, lane, M);
    v16h bf = load_b_frag(w_base, ldw, k0, lane, Nout - n0);
    acc = __builtin_amdgcn_wmma_f32_16x16x32_f16(false, af, false, bf,
                                                 (short)0, acc, false, false);
  }
  int col = lane & 15, rbase = (lane >> 4) << 3;
  int n = n0 + col;
  if (n >= Nout) return;
  float bval = 0.f;
  if (HAS_BIAS) bval = bias[(size_t)node * bStride + n];
#pragma unroll
  for (int r = 0; r < 8; ++r) {
    int m = m0 + rbase + r;
    if (m < M) {
      float v = acc[r] + bval;
      if (RELU) v = v > 0.f ? v : 0.f;
      size_t oidx = (size_t)node * oStride + (size_t)m * ldo + n;
      if (OUT_HALF) ((_Float16*)Out)[oidx] = (_Float16)v;
      else          ((float*)Out)[oidx]    = v;
    }
  }
}

// ---------------------------------------------------------------------------
// Attention: per (node, head). Wave w handles query rows [32w, 32w+32) as two
// 16-row groups. Scores (16x256) via 16 WMMAs live in registers; softmax with
// 16-lane shuffle reductions; probs staged fp16 in LDS; O = P@V via WMMA.
// ---------------------------------------------------------------------------
__global__ __launch_bounds__(256) void attn_kernel(const _Float16* __restrict__ qkv,
                                                   _Float16* __restrict__ oh) {
  __shared__ __align__(32) _Float16 probs[8][16][SEQ_S];   // 64 KB
  int node = blockIdx.x, head = blockIdx.y;
  int wave = threadIdx.x >> 5, lane = threadIdx.x & 31;
  const int ldq = 3 * DMODEL;
  const _Float16* base = qkv + (size_t)node * SEQ_S * ldq;
  int qoff = head * DHEAD;
  int koffH = DMODEL + head * DHEAD;
  int voff = 2 * DMODEL + head * DHEAD;
  const float scale = 0.17677669529663687f;   // 1/sqrt(32)
  int col = lane & 15, rbase = (lane >> 4) << 3;

  for (int g = 0; g < 2; ++g) {
    int m0 = wave * 32 + g * 16;
    v16h qf = load_a_frag(base + qoff, ldq, m0, 0, lane, SEQ_S);
    v8f acc[16];
#pragma unroll
    for (int j = 0; j < 16; ++j) {
      // B = K^T (32x16): column n == row (16j+n) of K, contiguous in memory
      v16h kf = load_b_frag(base + (size_t)(j * 16) * ldq + koffH, ldq, 0, lane, 16);
      v8f z = {};
      acc[j] = __builtin_amdgcn_wmma_f32_16x16x32_f16(false, qf, false, kf,
                                                      (short)0, z, false, false);
    }
    float mx[8], sm[8];
#pragma unroll
    for (int r = 0; r < 8; ++r) mx[r] = -1e30f;
#pragma unroll
    for (int j = 0; j < 16; ++j)
#pragma unroll
      for (int r = 0; r < 8; ++r) {
        float v = acc[j][r] * scale;
        acc[j][r] = v;
        mx[r] = fmaxf(mx[r], v);
      }
#pragma unroll
    for (int mk = 8; mk >= 1; mk >>= 1)
#pragma unroll
      for (int r = 0; r < 8; ++r) mx[r] = fmaxf(mx[r], __shfl_xor(mx[r], mk, 32));
#pragma unroll
    for (int r = 0; r < 8; ++r) sm[r] = 0.f;
#pragma unroll
    for (int j = 0; j < 16; ++j)
#pragma unroll
      for (int r = 0; r < 8; ++r) {
        float e = __expf(acc[j][r] - mx[r]);
        acc[j][r] = e;
        sm[r] += e;
      }
#pragma unroll
    for (int mk = 8; mk >= 1; mk >>= 1)
#pragma unroll
      for (int r = 0; r < 8; ++r) sm[r] += __shfl_xor(sm[r], mk, 32);
#pragma unroll
    for (int r = 0; r < 8; ++r) sm[r] = 1.0f / sm[r];
#pragma unroll
    for (int j = 0; j < 16; ++j)
#pragma unroll
      for (int r = 0; r < 8; ++r)
        probs[wave][rbase + r][j * 16 + col] = (_Float16)(acc[j][r] * sm[r]);

    // O tile = P(16x256) @ V(256x32); same-wave LDS store->load is in-order
#pragma unroll
    for (int nt = 0; nt < 2; ++nt) {
      v8f oacc = {};
      for (int kt = 0; kt < 8; ++kt) {
        v16h af = load_a_frag(&probs[wave][0][0], SEQ_S, 0, kt * 32, lane, 16);
        int kb = kt * 32 + ((lane >> 4) << 4);
        const _Float16* vp = base + (size_t)kb * ldq + voff + nt * 16 + col;
        v16h vf;
#pragma unroll
        for (int i = 0; i < 16; ++i) vf[i] = vp[(size_t)i * ldq];
        oacc = __builtin_amdgcn_wmma_f32_16x16x32_f16(false, af, false, vf,
                                                      (short)0, oacc, false, false);
      }
      _Float16* op = oh + (size_t)node * SEQ_S * DMODEL +
                     (size_t)m0 * DMODEL + head * DHEAD + nt * 16 + col;
#pragma unroll
      for (int r = 0; r < 8; ++r)
        op[(size_t)(rbase + r) * DMODEL] = (_Float16)oacc[r];
    }
  }
}

// ---------------------------------------------------------------------------
// residual + post-LN over D=128; updates f32 residual base and f16 activations
// ---------------------------------------------------------------------------
__global__ __launch_bounds__(128) void ln_kernel(const float* __restrict__ t,
                                                 float* __restrict__ xf,
                                                 _Float16* __restrict__ xh,
                                                 const float* __restrict__ gamma,
                                                 const float* __restrict__ beta,
                                                 long gStride) {
  __shared__ float red[8];
  int row = blockIdx.x;            // 0 .. N*S-1
  int node = row >> 8;             // S = 256
  int d = threadIdx.x;
  size_t idx = (size_t)row * DMODEL + d;
  float v = xf[idx] + t[idx];
  float s = v;
#pragma unroll
  for (int mk = 16; mk >= 1; mk >>= 1) s += __shfl_xor(s, mk, 32);
  if ((d & 31) == 0) red[d >> 5] = s;
  __syncthreads();
  float mean = (red[0] + red[1] + red[2] + red[3]) * (1.f / 128.f);
  float dv = v - mean;
  float q = dv * dv;
#pragma unroll
  for (int mk = 16; mk >= 1; mk >>= 1) q += __shfl_xor(q, mk, 32);
  if ((d & 31) == 0) red[4 + (d >> 5)] = q;
  __syncthreads();
  float var = (red[4] + red[5] + red[6] + red[7]) * (1.f / 128.f);
  float y = dv * rsqrtf(var + 1e-5f);
  y = y * gamma[(size_t)node * gStride + d] + beta[(size_t)node * gStride + d];
  xf[idx] = y;
  xh[idx] = (_Float16)y;
}

// ---------------------------------------------------------------------------
// misc elementwise kernels
// ---------------------------------------------------------------------------
__global__ void cvt_kernel(const float* __restrict__ in, _Float16* __restrict__ out, long n) {
  long i = (long)blockIdx.x * blockDim.x + threadIdx.x;
  long stride = (long)gridDim.x * blockDim.x;
  for (; i < n; i += stride) out[i] = (_Float16)in[i];
}

__global__ void pack_x_kernel(const float* __restrict__ x, float* __restrict__ xf,
                              _Float16* __restrict__ xh) {
  long total = (long)NNODES * SEQ_S * DMODEL;
  long i = (long)blockIdx.x * blockDim.x + threadIdx.x;
  long stride = (long)gridDim.x * blockDim.x;
  for (; i < total; i += stride) {
    int d = (int)(i % DMODEL);
    long t = i / DMODEL;
    int r = (int)(t % SEQ_S);      // batch
    int n = (int)(t / SEQ_S);      // node
    float v = x[(size_t)r * NNODES * DMODEL + (size_t)n * DMODEL + d];
    xf[i] = v;
    xh[i] = (_Float16)v;
  }
}

// ---------------------------------------------------------------------------
// GAT pieces
// ---------------------------------------------------------------------------
__device__ __forceinline__ unsigned fenc(float f) {
  unsigned u = __float_as_uint(f);
  return (u & 0x80000000u) ? ~u : (u | 0x80000000u);
}
__device__ __forceinline__ float fdec(unsigned u) {
  unsigned v = (u & 0x80000000u) ? (u & 0x7FFFFFFFu) : ~u;
  return __uint_as_float(v);
}

__global__ __launch_bounds__(32) void asd_kernel(const float* __restrict__ h,
                                                 const float* __restrict__ att_src,
                                                 const float* __restrict__ att_dst,
                                                 float* __restrict__ a_s,
                                                 float* __restrict__ a_d) {
  int row = blockIdx.x;
  int t = threadIdx.x;
  const float* hp = h + (size_t)row * GOUT;
  float s = hp[t] * att_src[t] + hp[t + 32] * att_src[t + 32];
  float d = hp[t] * att_dst[t] + hp[t + 32] * att_dst[t + 32];
#pragma unroll
  for (int mk = 16; mk >= 1; mk >>= 1) {
    s += __shfl_xor(s, mk, 32);
    d += __shfl_xor(d, mk, 32);
  }
  if (t == 0) { a_s[row] = s; a_d[row] = d; }
}

__global__ void gat_init_kernel(float* __restrict__ denom, unsigned* __restrict__ emax,
                                float* __restrict__ gout) {
  long total = (long)NROWS * GOUT;
  long i = (long)blockIdx.x * blockDim.x + threadIdx.x;
  long stride = (long)gridDim.x * blockDim.x;
  for (; i < total; i += stride) {
    gout[i] = 0.f;
    if (i < NROWS) { denom[i] = 0.f; emax[i] = 0u; }
  }
}

__global__ void edge_pass1(const int* __restrict__ ei, const float* __restrict__ a_s,
                           const float* __restrict__ a_d, float* __restrict__ ebuf,
                           unsigned* __restrict__ emax) {
  int i = blockIdx.x * 256 + threadIdx.x;
  if (i >= SEQ_S * ETOT) return;
  int b = i / ETOT, j = i % ETOT;
  int src = (j < NEDGE) ? ei[j] : (j - NEDGE);
  int dst = (j < NEDGE) ? ei[NEDGE + j] : (j - NEDGE);
  float e = a_s[b * NNODES + src] + a_d[b * NNODES + dst];
  e = (e > 0.f) ? e : 0.2f * e;            // leaky_relu(0.2)
  ebuf[i] = e;
  atomicMax(&emax[b * NNODES + dst], fenc(e));
}

__global__ void edge_pass2(const int* __restrict__ ei, float* __restrict__ ebuf,
                           const unsigned* __restrict__ emax, float* __restrict__ denom) {
  int i = blockIdx.x * 256 + threadIdx.x;
  if (i >= SEQ_S * ETOT) return;
  int b = i / ETOT, j = i % ETOT;
  int dst = (j < NEDGE) ? ei[NEDGE + j] : (j - NEDGE);
  float w = __expf(ebuf[i] - fdec(emax[b * NNODES + dst]));
  ebuf[i] = w;
  atomicAdd(&denom[b * NNODES + dst], w);
}

__global__ __launch_bounds__(64) void edge_pass3(const int* __restrict__ ei,
                                                 const float* __restrict__ ebuf,
                                                 const float* __restrict__ denom,
                                                 const float* __restrict__ h,
                                                 float* __restrict__ gout) {
  int bj = blockIdx.x;
  int b = bj / ETOT, j = bj % ETOT;
  int src = (j < NEDGE) ? ei[j] : (j - NEDGE);
  int dst = (j < NEDGE) ? ei[NEDGE + j] : (j - NEDGE);
  float alpha = ebuf[bj] / (denom[b * NNODES + dst] + 1e-16f);
  int c = threadIdx.x;
  atomicAdd(&gout[((size_t)b * NNODES + dst) * GOUT + c],
            alpha * h[((size_t)b * NNODES + src) * GOUT + c]);
}

__global__ __launch_bounds__(32) void pred_kernel(const float* __restrict__ gout,
                                                  const float* __restrict__ gat_b,
                                                  const float* __restrict__ pred_W,
                                                  const float* __restrict__ pred_b,
                                                  float* __restrict__ out) {
  int row = blockIdx.x;                    // b*N + n
  int t = threadIdx.x;
  float g0 = gout[(size_t)row * GOUT + t] + gat_b[t];
  float g1 = gout[(size_t)row * GOUT + t + 32] + gat_b[t + 32];
  float p0 = g0 * pred_W[t] + g1 * pred_W[t + 32];
  float p1 = g0 * pred_W[GOUT + t] + g1 * pred_W[GOUT + t + 32];
#pragma unroll
  for (int mk = 16; mk >= 1; mk >>= 1) {
    p0 += __shfl_xor(p0, mk, 32);
    p1 += __shfl_xor(p1, mk, 32);
  }
  if (t == 0) {
    out[(size_t)row * 2 + 0] = p0 + pred_b[0];
    out[(size_t)row * 2 + 1] = p1 + pred_b[1];
  }
}

// ---------------------------------------------------------------------------
extern "C" void kernel_launch(void* const* d_in, const int* in_sizes, int n_in,
                              void* d_out, int out_size, void* d_ws, size_t ws_size,
                              hipStream_t stream) {
  (void)in_sizes; (void)n_in; (void)out_size; (void)ws_size;
  const float* x       = (const float*)d_in[0];
  const int*   ei      = (const int*)d_in[1];
  const float* Wqkv    = (const float*)d_in[2];
  const float* bqkv    = (const float*)d_in[3];
  const float* Wo      = (const float*)d_in[4];
  const float* bo      = (const float*)d_in[5];
  const float* g1      = (const float*)d_in[6];
  const float* be1     = (const float*)d_in[7];
  const float* fW1     = (const float*)d_in[8];
  const float* fb1     = (const float*)d_in[9];
  const float* fW2     = (const float*)d_in[10];
  const float* fb2     = (const float*)d_in[11];
  const float* g2      = (const float*)d_in[12];
  const float* be2     = (const float*)d_in[13];
  const float* fc_W    = (const float*)d_in[14];
  const float* fc_b    = (const float*)d_in[15];
  const float* gat_W   = (const float*)d_in[16];
  const float* att_src = (const float*)d_in[17];
  const float* att_dst = (const float*)d_in[18];
  const float* gat_b   = (const float*)d_in[19];
  const float* pred_W  = (const float*)d_in[20];
  const float* pred_b  = (const float*)d_in[21];
  float* out = (float*)d_out;

  char* ws = (char*)d_ws;
  size_t off = 0;
  auto alloc = [&](size_t bytes) -> void* {
    void* p = ws + off;
    off += (bytes + 255) & ~(size_t)255;
    return p;
  };

  const long nWqkv = (long)NNODES * NLAYERS * 3 * DMODEL * DMODEL;  // 19.66M
  const long nWo   = (long)NNODES * NLAYERS * DMODEL * DMODEL;      // 6.55M
  const long nAct  = (long)NNODES * SEQ_S * DMODEL;                 // 6.55M

  _Float16* wqkv_h = (_Float16*)alloc((size_t)nWqkv * 2);
  _Float16* wo_h   = (_Float16*)alloc((size_t)nWo * 2);
  _Float16* fw1_h  = (_Float16*)alloc((size_t)nWo * 2);
  _Float16* fw2_h  = (_Float16*)alloc((size_t)nWo * 2);
  _Float16* fcW_h  = (_Float16*)alloc((size_t)FOUT * DMODEL * 2);
  _Float16* gatW_h = (_Float16*)alloc((size_t)GOUT * FOUT * 2);
  _Float16* xh     = (_Float16*)alloc((size_t)nAct * 2);
  float*    xf     = (float*)alloc((size_t)nAct * 4);
  _Float16* qkvh   = (_Float16*)alloc((size_t)NNODES * SEQ_S * 3 * DMODEL * 2);
  float*    tmpf   = (float*)qkvh;   // alias: qkv dead once attention has run
  _Float16* oh     = (_Float16*)alloc((size_t)nAct * 2);   // attn out / FFN hidden
  _Float16* featH  = (_Float16*)alloc((size_t)NROWS * FOUT * 2);
  float*    hbuf   = (float*)alloc((size_t)NROWS * GOUT * 4);
  float*    a_s    = (float*)alloc((size_t)NROWS * 4);
  float*    a_d    = (float*)alloc((size_t)NROWS * 4);
  unsigned* emax   = (unsigned*)alloc((size_t)NROWS * 4);
  float*    denom  = (float*)alloc((size_t)NROWS * 4);
  float*    ebuf   = (float*)alloc((size_t)SEQ_S * ETOT * 4);
  float*    gout   = (float*)alloc((size_t)NROWS * GOUT * 4);

  // ---- weight fp32 -> fp16, activation pack ----
  cvt_kernel<<<4096, 256, 0, stream>>>(Wqkv, wqkv_h, nWqkv);
  cvt_kernel<<<4096, 256, 0, stream>>>(Wo,   wo_h,   nWo);
  cvt_kernel<<<4096, 256, 0, stream>>>(fW1,  fw1_h,  nWo);
  cvt_kernel<<<4096, 256, 0, stream>>>(fW2,  fw2_h,  nWo);
  cvt_kernel<<<64,   256, 0, stream>>>(fc_W, fcW_h,  (long)FOUT * DMODEL);
  cvt_kernel<<<16,   256, 0, stream>>>(gat_W, gatW_h, (long)GOUT * FOUT);
  pack_x_kernel<<<4096, 256, 0, stream>>>(x, xf, xh);

  // ---- per-node transformer encoder, 2 layers ----
  for (int l = 0; l < NLAYERS; ++l) {
    gemm_node<false, true, true><<<dim3(NNODES, 8, 6), 256, 0, stream>>>(
        xh, (long)SEQ_S * DMODEL, DMODEL,
        wqkv_h + (size_t)l * 3 * DMODEL * DMODEL, (long)NLAYERS * 3 * DMODEL * DMODEL, DMODEL,
        bqkv + (size_t)l * 3 * DMODEL, (long)NLAYERS * 3 * DMODEL,
        qkvh, (long)SEQ_S * 3 * DMODEL, 3 * DMODEL,
        SEQ_S, 3 * DMODEL, DMODEL);

    attn_kernel<<<dim3(NNODES, HEADS_N), 256, 0, stream>>>(qkvh, oh);

    gemm_node<false, false, true><<<dim3(NNODES, 8, 2), 256, 0, stream>>>(
        oh, (long)SEQ_S * DMODEL, DMODEL,
        wo_h + (size_t)l * DMODEL * DMODEL, (long)NLAYERS * DMODEL * DMODEL, DMODEL,
        bo + (size_t)l * DMODEL, (long)NLAYERS * DMODEL,
        tmpf, (long)SEQ_S * DMODEL, DMODEL, SEQ_S, DMODEL, DMODEL);

    ln_kernel<<<NNODES * SEQ_S, 128, 0, stream>>>(
        tmpf, xf, xh, g1 + (size_t)l * DMODEL, be1 + (size_t)l * DMODEL,
        (long)NLAYERS * DMODEL);

    gemm_node<true, true, true><<<dim3(NNODES, 8, 2), 256, 0, stream>>>(
        xh, (long)SEQ_S * DMODEL, DMODEL,
        fw1_h + (size_t)l * DMODEL * DMODEL, (long)NLAYERS * DMODEL * DMODEL, DMODEL,
        fb1 + (size_t)l * DMODEL, (long)NLAYERS * DMODEL,
        oh, (long)SEQ_S * DMODEL, DMODEL, SEQ_S, DMODEL, DMODEL);

    gemm_node<false, false, true><<<dim3(NNODES, 8, 2), 256, 0, stream>>>(
        oh, (long)SEQ_S * DMODEL, DMODEL,
        fw2_h + (size_t)l * DMODEL * DMODEL, (long)NLAYERS * DMODEL * DMODEL, DMODEL,
        fb2 + (size_t)l * DMODEL, (long)NLAYERS * DMODEL,
        tmpf, (long)SEQ_S * DMODEL, DMODEL, SEQ_S, DMODEL, DMODEL);

    ln_kernel<<<NNODES * SEQ_S, 128, 0, stream>>>(
        tmpf, xf, xh, g2 + (size_t)l * DMODEL, be2 + (size_t)l * DMODEL,
        (long)NLAYERS * DMODEL);
  }

  // ---- shared fc: feat(flat row = node*256 + batch) = t_out @ fc_W^T + b ----
  gemm_node<false, true, true><<<dim3(NNODES, 8, 1), 256, 0, stream>>>(
      xh, (long)SEQ_S * DMODEL, DMODEL,
      fcW_h, 0L, DMODEL,
      fc_b, 0L,
      featH, (long)SEQ_S * FOUT, FOUT, SEQ_S, FOUT, DMODEL);

  // ---- GAT: torch reshape scramble means batch b uses flat rows b*200..+199
  //      (contiguous in featH) -> batched GEMM h = feat @ gat_W^T ----
  gemm_node<false, false, false><<<dim3(SEQ_S, 7, 1), 256, 0, stream>>>(
      featH, (long)NNODES * FOUT, FOUT,
      gatW_h, 0L, FOUT,
      (const float*)nullptr, 0L,
      hbuf, (long)NNODES * GOUT, GOUT, NNODES, GOUT, FOUT);

  asd_kernel<<<NROWS, 32, 0, stream>>>(hbuf, att_src, att_dst, a_s, a_d);
  gat_init_kernel<<<4096, 256, 0, stream>>>(denom, emax, gout);
  {
    int total = SEQ_S * ETOT;
    int blocks = (total + 255) / 256;
    edge_pass1<<<blocks, 256, 0, stream>>>(ei, a_s, a_d, ebuf, emax);
    edge_pass2<<<blocks, 256, 0, stream>>>(ei, ebuf, emax, denom);
    edge_pass3<<<total, 64, 0, stream>>>(ei, ebuf, denom, hbuf, gout);
  }
  pred_kernel<<<NROWS, 32, 0, stream>>>(gout, gat_b, pred_W, pred_b, out);
}
